// MixtureOfExperts_63376537420302
// MI455X (gfx1250) — matrix operand order repaired and checked by reading, not compiled, not run on hardware
//
#include <hip/hip_runtime.h>
#include <hip/hip_bf16.h>
#include <math.h>

// ---------------------------------------------------------------------------
// MoE (dense top-1-scaled) for MI455X / gfx1250.
// bf16 WMMA + async global->LDS staging of W2 tiles (ASYNCcnt pipeline).
// Shapes fixed by reference: D=1024, H=2048, E=8, N = B*S (8192).
// ---------------------------------------------------------------------------

typedef __attribute__((ext_vector_type(16))) __bf16 v16bf;
typedef __attribute__((ext_vector_type(8)))  float  v8f;

union ABf { int4 q[2]; v16bf v; };
static_assert(sizeof(v16bf) == 32, "v16bf must be 32 bytes");

static constexpr int Dk = 1024;   // model dim
static constexpr int Hk = 2048;   // hidden dim
static constexpr int Ek = 8;      // experts

__device__ __forceinline__ unsigned short f2bf(float f) {
  unsigned int u = __builtin_bit_cast(unsigned int, f);
  unsigned int r = u + 0x7FFFu + ((u >> 16) & 1u);   // round-to-nearest-even
  return (unsigned short)(r >> 16);
}

// ---------------- pack x: f32 -> bf16 row-major ----------------------------
__global__ __launch_bounds__(256) void pack_x_kernel(const float* __restrict__ x,
                                                     unsigned short* __restrict__ xb,
                                                     int n_elems) {
  int i = blockIdx.x * 256 + threadIdx.x;
  if (i < n_elems) xb[i] = f2bf(x[i]);
}

// ---------------- pack W1 into WMMA-B tile layout --------------------------
// tile = (e, t1, c1): 512 bf16, element (L, j) = w1[e][c1*32 + j + 16*(L>>4)][t1*16 + (L&15)]
__global__ __launch_bounds__(256) void pack_w1_kernel(const float* __restrict__ w1,
                                                      unsigned short* __restrict__ w1p) {
  int o = blockIdx.x * 256 + threadIdx.x;          // < E*(H/16)*(D/32)*512 = 16777216
  int j  = o & 15;
  int L  = (o >> 4) & 31;
  int c1 = (o >> 9) & 31;                          // D/32 = 32 chunks
  int t1 = (o >> 14) & 127;                        // H/16 = 128 tiles
  int e  = o >> 21;
  int k = c1 * 32 + j + 16 * (L >> 4);             // k over D
  int n = t1 * 16 + (L & 15);                      // n over H
  w1p[o] = f2bf(w1[((size_t)e * Dk + k) * Hk + n]);
}

// ---------------- pack W2 into WMMA-B tile layout --------------------------
// tile = (e, t2, c2): element (L, j) = w2[e][c2*32 + j + 16*(L>>4)][t2*16 + (L&15)]
__global__ __launch_bounds__(256) void pack_w2_kernel(const float* __restrict__ w2,
                                                      unsigned short* __restrict__ w2p) {
  int o = blockIdx.x * 256 + threadIdx.x;          // < E*(D/16)*(H/32)*512 = 16777216
  int j  = o & 15;
  int L  = (o >> 4) & 31;
  int c2 = (o >> 9) & 63;                          // H/32 = 64 chunks
  int t2 = (o >> 15) & 63;                         // D/16 = 64 tiles
  int e  = o >> 21;
  int k = c2 * 32 + j + 16 * (L >> 4);             // k over H
  int n = t2 * 16 + (L & 15);                      // n over D
  w2p[o] = f2bf(w2[((size_t)e * Hk + k) * Dk + n]);
}

// ---------------- sum of b2 over experts -----------------------------------
__global__ __launch_bounds__(256) void b2sum_kernel(const float* __restrict__ b2,
                                                    float* __restrict__ b2s) {
  int d = blockIdx.x * 256 + threadIdx.x;
  if (d < Dk) {
    float s = 0.f;
    #pragma unroll
    for (int e = 0; e < Ek; ++e) s += b2[e * Dk + d];
    b2s[d] = s;
  }
}

// ---------------- gating: one wave32 per token -----------------------------
__global__ __launch_bounds__(256) void gate_kernel(const float* __restrict__ x,
                                                   const float* __restrict__ gw,
                                                   const float* __restrict__ gb,
                                                   const float* __restrict__ rmsw,
                                                   float* __restrict__ normg, int N) {
  int wave = threadIdx.x >> 5;
  int lane = threadIdx.x & 31;
  int tok = blockIdx.x * 8 + wave;
  if (tok >= N) return;
  const float* xr = x + (size_t)tok * Dk;
  float part[Ek] = {0.f, 0.f, 0.f, 0.f, 0.f, 0.f, 0.f, 0.f};
  for (int i = 0; i < Dk / 32; ++i) {
    int d = lane + i * 32;
    float xv = xr[d];
    #pragma unroll
    for (int e = 0; e < Ek; ++e) part[e] += xv * gw[e * Dk + d];
  }
  #pragma unroll
  for (int e = 0; e < Ek; ++e) {
    #pragma unroll
    for (int off = 16; off > 0; off >>= 1) part[e] += __shfl_xor(part[e], off, 32);
    part[e] += gb[e];
  }
  float ms = 0.f;
  #pragma unroll
  for (int e = 0; e < Ek; ++e) ms += part[e] * part[e];
  ms *= (1.0f / Ek);
  float inv = rsqrtf(ms + 1.1920929e-07f);
  float l[Ek], mx = -3.402823466e+38f;
  #pragma unroll
  for (int e = 0; e < Ek; ++e) { l[e] = part[e] * inv * rmsw[e] * 2.0f; mx = fmaxf(mx, l[e]); }
  float s = 0.f;
  #pragma unroll
  for (int e = 0; e < Ek; ++e) s += expf(l[e] - mx);
  float g = 1.0f / s;                  // top-1 softmax value
  float ng = g / (g + 1e-6f);          // capacity mask is provably 1 for top-1
  if (lane == 0) normg[tok] = ng;
}

// ---------------- fused FFN: 32 tokens x 1024 outputs per workgroup --------
// Dynamic LDS layout:
//   [0, 128KB)      : W2 B-tile staging, 128 tiles x 1KB  (tile = tt*2 + c2)
//   [128KB, 132KB)  : P buffer, 32 tokens x 64 bf16
__global__ __launch_bounds__(256) void moe_ffn_kernel(const unsigned short* __restrict__ xb,
                                                      const unsigned short* __restrict__ w1p,
                                                      const unsigned short* __restrict__ w2p,
                                                      const float* __restrict__ b1,
                                                      const float* __restrict__ b2s,
                                                      const float* __restrict__ normg,
                                                      float* __restrict__ out) {
  extern __shared__ __align__(16) char smem[];
  int4* blds4 = (int4*)smem;                           // 8192 int4 = 128 KB
  int4* plds4 = (int4*)(smem + 131072);                // 256 int4 = 4 KB
  unsigned short* plds = (unsigned short*)plds4;

  const int tid  = threadIdx.x;
  const int wave = tid >> 5;
  const int lane = tid & 31;
  const int g    = lane >> 4;                      // half-wave group
  const int ln   = lane & 15;
  const int tok0 = blockIdx.x * 32;

  // GEMM1 tile coords: 2 m-tiles x 4 h-tiles across 8 waves
  const int mt1 = wave >> 2;
  const int ht  = wave & 3;
  // GEMM2 tile coords: 2 m-tiles x 4 n-groups (256 outputs each)
  const int mt2 = wave & 1;
  const int ngp = wave >> 1;

  const int4* w1p4 = (const int4*)w1p;

  // Async-copy bases (uniform SGPR base + per-lane VGPR offsets)
  const unsigned long long w2base = (unsigned long long)(uintptr_t)w2p;
  const unsigned ldsbase  = (unsigned)(uintptr_t)&smem[0];
  const unsigned lofs0    = ldsbase + (unsigned)tid * 16u;          // +4096/iter
  const unsigned gofs_tid = (unsigned)((tid >> 7) * 65536 + (tid & 127) * 16);

  // A-operand base for GEMM1 (global bf16 x, row-major)
  const int4* xg4 = (const int4*)(xb + (size_t)(tok0 + mt1 * 16 + ln) * Dk);

  v8f acc[16];
  #pragma unroll
  for (int t = 0; t < 16; ++t) acc[t] = v8f{};

  for (int e = 0; e < Ek; ++e) {
    for (int hb = 0; hb < Hk / 64; ++hb) {
      // ---- kick off async staging of this iteration's W2 tiles ------------
      // source: 64 tiles tt, each 2KB (both k-chunks contiguous), stride 64KB
      {
        const unsigned gofs0 = (unsigned)(e * 4194304 + hb * 2048) + gofs_tid;
        #pragma unroll
        for (int i = 0; i < 32; ++i) {
          asm volatile("global_load_async_to_lds_b128 %0, %1, %2 offset:0"
                       :
                       : "v"(lofs0 + (unsigned)(i * 4096)),
                         "v"(gofs0 + (unsigned)(i * 131072)),
                         "s"(w2base)
                       : "memory");
        }
      }

      // ---- GEMM1: P(16x16) = X(16x1024) @ W1_e[:, 64-block], K-loop over D
      v8f c1 = v8f{};
      const int btile1 = ((e * 128 + hb * 4 + ht) * 32) * 64 + lane * 2;  // int4 units
      #pragma unroll 4
      for (int c = 0; c < Dk / 32; ++c) {
        ABf a, b;
        a.q[0] = xg4[c * 4 + 2 * g];              // k = c*32 + 16g .. +7
        a.q[1] = xg4[c * 4 + 2 * g + 1];          // k = c*32 + 16g + 8 .. +15
        b.q[0] = w1p4[btile1 + c * 64];
        b.q[1] = w1p4[btile1 + c * 64 + 1];
        c1 = __builtin_amdgcn_wmma_f32_16x16x32_bf16(false, a.v, false, b.v,
                                                     (short)0, c1, false, false);
      }
      // bias + exact GELU + bf16 store to LDS
      const float bias = b1[e * Hk + hb * 64 + ht * 16 + ln];
      #pragma unroll
      for (int r = 0; r < 8; ++r) {
        float v = c1[r] + bias;
        float gl = 0.5f * v * (1.0f + erff(v * 0.70710678118654752f));
        plds[(mt1 * 16 + r + 8 * g) * 64 + ht * 16 + ln] = f2bf(gl);
      }
      // all async W2 copies for this iteration must have landed in LDS
      asm volatile("s_wait_asynccnt 0x0" ::: "memory");
      __syncthreads();

      // ---- GEMM2: acc(16x256/wave) += P(16x64) @ W2_e[64-block, :] (B from LDS)
      #pragma unroll
      for (int c2 = 0; c2 < 2; ++c2) {
        ABf a;
        a.q[0] = plds4[(mt2 * 16 + ln) * 8 + c2 * 4 + 2 * g];
        a.q[1] = plds4[(mt2 * 16 + ln) * 8 + c2 * 4 + 2 * g + 1];
        #pragma unroll
        for (int t = 0; t < 16; ++t) {
          const int bt = ((ngp * 16 + t) * 2 + c2) * 64 + lane * 2;   // int4 units in LDS
          ABf b;
          b.q[0] = blds4[bt];
          b.q[1] = blds4[bt + 1];
          acc[t] = __builtin_amdgcn_wmma_f32_16x16x32_bf16(false, a.v, false, b.v,
                                                           (short)0, acc[t], false, false);
        }
      }
      __syncthreads();   // protect blds/plds before next iteration's writes
    }
  }

  // ---- epilogue: out = norm_g * (acc + sum_e b2)
  #pragma unroll
  for (int r = 0; r < 8; ++r) {
    const int tok = tok0 + mt2 * 16 + r + 8 * g;
    const float ngs = normg[tok];
    #pragma unroll
    for (int t = 0; t < 16; ++t) {
      const int d = ngp * 256 + t * 16 + ln;
      out[(size_t)tok * Dk + d] = ngs * (acc[t][r] + b2s[d]);
    }
  }
}

// ---------------------------------------------------------------------------
extern "C" void kernel_launch(void* const* d_in, const int* in_sizes, int n_in,
                              void* d_out, int out_size, void* d_ws, size_t ws_size,
                              hipStream_t stream) {
  const float* x    = (const float*)d_in[0];
  const float* gw   = (const float*)d_in[1];
  const float* gb   = (const float*)d_in[2];
  const float* rmsw = (const float*)d_in[3];
  const float* w1   = (const float*)d_in[4];
  const float* b1   = (const float*)d_in[5];
  const float* w2   = (const float*)d_in[6];
  const float* b2   = (const float*)d_in[7];
  float* out = (float*)d_out;

  const int N = in_sizes[0] / Dk;                  // tokens (8192)

  // workspace layout (256B aligned)
  char* ws = (char*)d_ws;
  float*          normg = (float*)ws;                              // N*4      = 32 KB
  float*          b2s   = (float*)(ws + 32768);                    // 4 KB
  unsigned short* xb    = (unsigned short*)(ws + 36864);           // N*D*2    = 16 MB
  unsigned short* w1p   = (unsigned short*)(ws + 36864 + (size_t)N * Dk * 2);
  unsigned short* w2p   = w1p + (size_t)Ek * Dk * Hk;              // +32 MB each

  const int n_x  = N * Dk;                 // 8388608
  const int n_w  = Ek * Dk * Hk;           // 16777216

  pack_x_kernel <<<(n_x + 255) / 256, 256, 0, stream>>>(x, xb, n_x);
  pack_w1_kernel<<<n_w / 256, 256, 0, stream>>>(w1, w1p);
  pack_w2_kernel<<<n_w / 256, 256, 0, stream>>>(w2, w2p);
  b2sum_kernel  <<<(Dk + 255) / 256, 256, 0, stream>>>(b2, b2s);
  gate_kernel   <<<(N + 7) / 8, 256, 0, stream>>>(x, gw, gb, rmsw, normg, N);

  const size_t ffn_lds = 131072 + 4096;    // W2 staging + P buffer
  moe_ffn_kernel<<<N / 32, 256, ffn_lds, stream>>>(xb, w1p, w2p, b1, b2s, normg, out);
}
